// GATv2ActorCritic_13185549598931
// MI455X (gfx1250) — matrix operand order repaired and checked by reading, not compiled, not run on hardware
//
#include <hip/hip_runtime.h>
#include <hip/hip_bf16.h>

typedef __attribute__((ext_vector_type(16))) _Float16 v16h;
typedef __attribute__((ext_vector_type(8)))  float    v8f;

#define NEG_SLOPE 0.2f

// ---------------------------------------------------------------------------
// helpers
// ---------------------------------------------------------------------------
__device__ __forceinline__ void atomicMaxF(float* addr, float val) {
  // works with addr initialized to -inf (0xFF800000)
  if (!(val < 0.0f)) atomicMax((int*)addr, __float_as_int(val));
  else               atomicMin((unsigned int*)addr, __float_as_uint(val));
}

__global__ void fill_f32(float* __restrict__ p, float v, int n) {
  int i = blockIdx.x * blockDim.x + threadIdx.x;
  if (i < n) p[i] = v;
}

// ---------------------------------------------------------------------------
// Pack a [KDIM,64] fp32 weight matrix into f16 WMMA B-fragment order:
//   out[((kc*4 + nt)*32 + lane)*16 + i]  <=  W[k*64 + col]
//   k = kc*32 + (i>>3)*16 + (lane>>4)*8 + (i&7),  col = nt*16 + (lane&15)
// (16-bit B 32x16 layout: lanes 0-15 hold K=0..7,16..23 of the chunk, 16-31 K=8..15,24..31)
// ---------------------------------------------------------------------------
__global__ void pack_w_frag(const float* __restrict__ W, _Float16* __restrict__ out,
                            int KDIM, int total) {
  int idx = blockIdx.x * blockDim.x + threadIdx.x;
  if (idx >= total) return;
  int i    = idx & 15;
  int lane = (idx >> 4) & 31;
  int nt   = (idx >> 9) & 3;
  int kc   = idx >> 11;
  int k    = kc * 32 + ((i >> 3) << 4) + (lane >> 4) * 8 + (i & 7);
  int col  = nt * 16 + (lane & 15);
  out[idx] = (k < KDIM) ? (_Float16)W[k * 64 + col] : (_Float16)0.0f;
}

// ---------------------------------------------------------------------------
// Node feature transform: XL = X @ Wl, XR = X @ Wr   (X:[n,KDIM], W packed f16)
// One wave per 16-row tile; 4 col-tiles of 16; f16 WMMA with fp32 accum.
// Fast path (full tile): unguarded float4 loads; B fragments are single
// 32-byte vector loads from the packed buffers.
// ---------------------------------------------------------------------------
template<int KDIM>
__global__ __launch_bounds__(128)
void node_gemm_dual(const float* __restrict__ X,
                    const _Float16* __restrict__ WfL,
                    const _Float16* __restrict__ WfR,
                    float* __restrict__ XL,
                    float* __restrict__ XR,
                    int n) {
  const int lane = threadIdx.x & 31;
  const int wave = threadIdx.x >> 5;
  const int tile = blockIdx.x * 4 + wave;       // 16-row output tile
  const int rowA = tile * 16 + (lane & 15);
  const int half = lane >> 4;
  constexpr int KCH = (KDIM + 31) / 32;
  const v16h* BL = (const v16h*)WfL;
  const v16h* BR = (const v16h*)WfR;

  v8f cl[4] = {};
  v8f cr[4] = {};
  const bool full = (tile * 16 + 16 <= n);

#pragma unroll
  for (int kc = 0; kc < KCH; ++kc) {
    float f[16];
#pragma unroll
    for (int i = 0; i < 16; ++i) f[i] = 0.0f;

    if (full) {
      // A fragment per lane = two contiguous 8-float runs:
      //   k in [kc*32 + half*8, +8) -> elements 0..7
      //   k in [kc*32 + 16 + half*8, +8) -> elements 8..15
      const float* base = X + (long long)rowA * KDIM + kc * 32 + half * 8;
      float4 q0 = *(const float4*)(base);
      float4 q1 = *(const float4*)(base + 4);
      f[0] = q0.x; f[1] = q0.y; f[2] = q0.z; f[3] = q0.w;
      f[4] = q1.x; f[5] = q1.y; f[6] = q1.z; f[7] = q1.w;
      if (kc * 32 + 16 + 8 <= KDIM) {          // folds after unroll
        float4 q2 = *(const float4*)(base + 16);
        float4 q3 = *(const float4*)(base + 20);
        f[8]  = q2.x; f[9]  = q2.y; f[10] = q2.z; f[11] = q2.w;
        f[12] = q3.x; f[13] = q3.y; f[14] = q3.z; f[15] = q3.w;
      }
    } else {
#pragma unroll
      for (int i = 0; i < 16; ++i) {
        int k = kc * 32 + ((i >> 3) << 4) + half * 8 + (i & 7);
        if (rowA < n && k < KDIM) f[i] = X[(long long)rowA * KDIM + k];
      }
    }

    v16h a;
#pragma unroll
    for (int i = 0; i < 16; ++i) a[i] = (_Float16)f[i];

#pragma unroll
    for (int nt = 0; nt < 4; ++nt) {
      v16h bl = BL[(kc * 4 + nt) * 32 + lane];
      v16h br = BR[(kc * 4 + nt) * 32 + lane];
      cl[nt] = __builtin_amdgcn_wmma_f32_16x16x32_f16(false, a, false, bl, (short)0, cl[nt], false, false);
      cr[nt] = __builtin_amdgcn_wmma_f32_16x16x32_f16(false, a, false, br, (short)0, cr[nt], false, false);
    }
  }

  // D layout: VGPR v, lanes 0-15 -> M=v, lanes 16-31 -> M=v+8; N = lane&15
  if (full) {
#pragma unroll
    for (int nt = 0; nt < 4; ++nt) {
      int col = nt * 16 + (lane & 15);
#pragma unroll
      for (int v = 0; v < 8; ++v) {
        int r = tile * 16 + v + half * 8;
        XL[(long long)r * 64 + col] = cl[nt][v];
        XR[(long long)r * 64 + col] = cr[nt][v];
      }
    }
  } else {
#pragma unroll
    for (int nt = 0; nt < 4; ++nt) {
      int col = nt * 16 + (lane & 15);
#pragma unroll
      for (int v = 0; v < 8; ++v) {
        int r = tile * 16 + v + half * 8;
        if (r < n) {
          XL[(long long)r * 64 + col] = cl[nt][v];
          XR[(long long)r * 64 + col] = cr[nt][v];
        }
      }
    }
  }
}

// ---------------------------------------------------------------------------
// Edge pass 1: logits[e,h] = att_h . LeakyReLU(xl[src] + xr[dst] + e@We)
// One wave per edge; lane handles channels (lane, lane+32); 16-lane head sums.
// Also atomic-max into m[dst,h].
// ---------------------------------------------------------------------------
__global__ __launch_bounds__(256)
void edge_logits(const int* __restrict__ src, const int* __restrict__ dst,
                 const float* __restrict__ eattr,
                 const float* __restrict__ XL, const float* __restrict__ XR,
                 const float* __restrict__ We,   // [4,64]
                 const float* __restrict__ att,  // [64] (h*16+c)
                 float* __restrict__ logits,     // [E,4]
                 float* __restrict__ m,          // [N,4], init -inf
                 int E_) {
  __shared__ float sWe[256];
  __shared__ float sAtt[64];
  int tid = threadIdx.x;
  if (tid < 256) sWe[tid] = We[tid];
  if (tid < 64)  sAtt[tid] = att[tid];
  __syncthreads();

  int wave = tid >> 5, lane = tid & 31;
  int e = blockIdx.x * 8 + wave;
  if (e >= E_) return;

  int s = src[e], d = dst[e];
  float4 ea = ((const float4*)eattr)[e];

  float acc0, acc1;
#pragma unroll
  for (int t = 0; t < 2; ++t) {
    int ch = lane + 32 * t;
    float ef = ea.x * sWe[ch] + ea.y * sWe[64 + ch] + ea.z * sWe[128 + ch] + ea.w * sWe[192 + ch];
    float sv = XL[(long long)s * 64 + ch] + XR[(long long)d * 64 + ch] + ef;
    sv = sv > 0.0f ? sv : NEG_SLOPE * sv;
    float v = sv * sAtt[ch];
    if (t == 0) acc0 = v; else acc1 = v;
  }
  // reduce within the two 16-lane groups
#pragma unroll
  for (int off = 1; off < 16; off <<= 1) {
    acc0 += __shfl_xor(acc0, off, 32);
    acc1 += __shfl_xor(acc1, off, 32);
  }
  if ((lane & 15) == 0) {
    int h0 = lane >> 4;  // lanes 0-15 -> heads {0,2}; lanes 16-31 -> heads {1,3}
    logits[(long long)e * 4 + h0]     = acc0;
    logits[(long long)e * 4 + h0 + 2] = acc1;
    atomicMaxF(&m[(long long)d * 4 + h0],     acc0);
    atomicMaxF(&m[(long long)d * 4 + h0 + 2], acc1);
  }
}

// ---------------------------------------------------------------------------
// Edge pass 2: ex = exp(logit - m[dst]); denom[dst] += ex (in place over logits)
// ---------------------------------------------------------------------------
__global__ __launch_bounds__(256)
void edge_exp(const int* __restrict__ dst,
              float* __restrict__ logits_ex,   // [E,4], in-place
              const float* __restrict__ m,     // [N,4]
              float* __restrict__ denom,       // [N,4], init 0
              int EH) {
  int i = blockIdx.x * blockDim.x + threadIdx.x;
  if (i >= EH) return;
  int e = i >> 2, h = i & 3;
  int d = dst[e];
  float ex = __expf(logits_ex[i] - m[(long long)d * 4 + h]);
  logits_ex[i] = ex;
  atomicAdd(&denom[(long long)d * 4 + h], ex);
}

// ---------------------------------------------------------------------------
// Edge pass 3: out[dst] += (ex/denom[dst]) * xl[src]   (wave per edge)
// ---------------------------------------------------------------------------
__global__ __launch_bounds__(256)
void edge_scatter(const int* __restrict__ src, const int* __restrict__ dst,
                  const float* __restrict__ XL,
                  const float* __restrict__ ex,     // [E,4]
                  const float* __restrict__ denom,  // [N,4]
                  float* __restrict__ out,          // [N,64], init 0
                  int E_) {
  int tid = threadIdx.x;
  int wave = tid >> 5, lane = tid & 31;
  int e = blockIdx.x * 8 + wave;
  if (e >= E_) return;
  int s = src[e], d = dst[e];
  int h0 = lane >> 4;
  float a0 = ex[(long long)e * 4 + h0]     / denom[(long long)d * 4 + h0];
  float a1 = ex[(long long)e * 4 + h0 + 2] / denom[(long long)d * 4 + h0 + 2];
  atomicAdd(&out[(long long)d * 64 + lane],      a0 * XL[(long long)s * 64 + lane]);
  atomicAdd(&out[(long long)d * 64 + lane + 32], a1 * XL[(long long)s * 64 + lane + 32]);
}

// ---------------------------------------------------------------------------
// y = elu(out + bias)
// ---------------------------------------------------------------------------
__global__ __launch_bounds__(256)
void finalize_elu(const float* __restrict__ acc, const float* __restrict__ bias,
                  float* __restrict__ y, int n64) {
  int i = blockIdx.x * blockDim.x + threadIdx.x;
  if (i >= n64) return;
  float v = acc[i] + bias[i & 63];
  y[i] = v > 0.0f ? v : (__expf(v) - 1.0f);
}

// ---------------------------------------------------------------------------
// Pool: gsum[batch[n]] += (out + bias); gcnt[batch[n]] += 1
// ---------------------------------------------------------------------------
__global__ __launch_bounds__(256)
void finalize_pool(const float* __restrict__ acc, const float* __restrict__ bias,
                   const int* __restrict__ batch,
                   float* __restrict__ gsum, float* __restrict__ gcnt, int n64) {
  int i = blockIdx.x * blockDim.x + threadIdx.x;
  if (i >= n64) return;
  int node = i >> 6, ch = i & 63;
  float v = acc[i] + bias[ch];
  int g = batch[node];
  atomicAdd(&gsum[g * 64 + ch], v);
  if (ch == 0) atomicAdd(&gcnt[g], 1.0f);
}

// ---------------------------------------------------------------------------
// Heads: per-graph MLPs (G=64, all tiny). One thread per graph.
// out layout: gmm [64*15] | proposal logits [64*10] | values [64]
// ---------------------------------------------------------------------------
__global__ __launch_bounds__(64)
void heads_kernel(const float* __restrict__ gsum, const float* __restrict__ gcnt,
                  const float* __restrict__ Wg1, const float* __restrict__ bg1,
                  const float* __restrict__ Wg2, const float* __restrict__ bg2,
                  const float* __restrict__ Wp,  const float* __restrict__ bp,
                  const float* __restrict__ Wc1, const float* __restrict__ bc1,
                  const float* __restrict__ Wc2, const float* __restrict__ bc2,
                  float* __restrict__ out) {
  int gi = threadIdx.x;
  if (gi >= 64) return;
  float cnt = fmaxf(gcnt[gi], 1.0f);
  float gv[64];
  for (int c = 0; c < 64; ++c) gv[c] = gsum[gi * 64 + c] / cnt;

  float h1[64];
  for (int j = 0; j < 64; ++j) {
    float a = bg1[j];
    for (int c = 0; c < 64; ++c) a += gv[c] * Wg1[c * 64 + j];
    h1[j] = fmaxf(a, 0.0f);
  }
  for (int k = 0; k < 15; ++k) {
    float a = bg2[k];
    for (int j = 0; j < 64; ++j) a += h1[j] * Wg2[j * 15 + k];
    out[gi * 15 + k] = a;
  }
  for (int k = 0; k < 10; ++k) {
    float a = bp[k];
    for (int c = 0; c < 64; ++c) a += gv[c] * Wp[c * 10 + k];
    out[960 + gi * 10 + k] = a;
  }
  float val = bc2[0];
  for (int j = 0; j < 64; ++j) {
    float a = bc1[j];
    for (int c = 0; c < 64; ++c) a += gv[c] * Wc1[c * 64 + j];
    val += fmaxf(a, 0.0f) * Wc2[j];
  }
  out[1600 + gi] = val;
}

// ---------------------------------------------------------------------------
// launcher
// ---------------------------------------------------------------------------
extern "C" void kernel_launch(void* const* d_in, const int* in_sizes, int n_in,
                              void* d_out, int out_size, void* d_ws, size_t ws_size,
                              hipStream_t stream) {
  const float* x     = (const float*)d_in[0];
  const int*   eidx  = (const int*)  d_in[1];
  const float* eattr = (const float*)d_in[2];
  const int*   batch = (const int*)  d_in[3];
  const float* Wl1 = (const float*)d_in[4];
  const float* Wr1 = (const float*)d_in[5];
  const float* We1 = (const float*)d_in[6];
  const float* at1 = (const float*)d_in[7];
  const float* b1  = (const float*)d_in[8];
  const float* Wl2 = (const float*)d_in[9];
  const float* Wr2 = (const float*)d_in[10];
  const float* We2 = (const float*)d_in[11];
  const float* at2 = (const float*)d_in[12];
  const float* b2  = (const float*)d_in[13];
  const float* Wg1 = (const float*)d_in[14];
  const float* bg1 = (const float*)d_in[15];
  const float* Wg2 = (const float*)d_in[16];
  const float* bg2 = (const float*)d_in[17];
  const float* Wp  = (const float*)d_in[18];
  const float* bp  = (const float*)d_in[19];
  const float* Wc1 = (const float*)d_in[20];
  const float* bc1 = (const float*)d_in[21];
  const float* Wc2 = (const float*)d_in[22];
  const float* bc2 = (const float*)d_in[23];

  const int N = in_sizes[0] / 16;
  const int E = in_sizes[2] / 4;
  const int* src = eidx;
  const int* dst = eidx + E;

  // workspace carve-up (floats)
  float* ws  = (float*)d_ws;
  const long long N64 = (long long)N * 64;
  const long long E4  = (long long)E * 4;
  float* XL  = ws;                     // N*64
  float* XR  = XL + N64;               // N*64
  float* ACC = XR + N64;               // N*64 (per-layer aggregation)
  float* Y   = ACC + N64;              // N*64 (post-elu layer-1 output)
  float* LG  = Y + N64;                // E*4  (logits, then ex, in place)
  float* M   = LG + E4;                // N*4
  float* DEN = M + (long long)N * 4;   // N*4
  float* GS  = DEN + (long long)N * 4; // 64*64
  float* GC  = GS + 64 * 64;           // 64
  _Float16* WPL = (_Float16*)(GC + 64);          // up to 2*2048 halfs
  _Float16* WPR = WPL + 2 * 2048;

  const int n64  = N * 64;
  const int n4   = N * 4;
  const int eh   = E * 4;
  const int gemmBlocks = (N + 63) / 64;     // 4 tiles of 16 rows per block
  const int edgeBlocks = (E + 7) / 8;       // 8 waves (edges) per block

  // ---------------- layer 1 ----------------
  {
    const int tot = 1 * 2048;                // KCH=1 for KDIM=16
    pack_w_frag<<<(tot + 255) / 256, 256, 0, stream>>>(Wl1, WPL, 16, tot);
    pack_w_frag<<<(tot + 255) / 256, 256, 0, stream>>>(Wr1, WPR, 16, tot);
  }
  node_gemm_dual<16><<<gemmBlocks, 128, 0, stream>>>(x, WPL, WPR, XL, XR, N);
  fill_f32<<<(n4  + 255) / 256, 256, 0, stream>>>(M,  -INFINITY, n4);
  fill_f32<<<(n4  + 255) / 256, 256, 0, stream>>>(DEN, 0.0f, n4);
  fill_f32<<<(n64 + 255) / 256, 256, 0, stream>>>(ACC, 0.0f, n64);
  edge_logits <<<edgeBlocks, 256, 0, stream>>>(src, dst, eattr, XL, XR, We1, at1, LG, M, E);
  edge_exp    <<<(eh + 255) / 256, 256, 0, stream>>>(dst, LG, M, DEN, eh);
  edge_scatter<<<edgeBlocks, 256, 0, stream>>>(src, dst, XL, LG, DEN, ACC, E);
  finalize_elu<<<(n64 + 255) / 256, 256, 0, stream>>>(ACC, b1, Y, n64);

  // ---------------- layer 2 ----------------
  {
    const int tot = 2 * 2048;                // KCH=2 for KDIM=64
    pack_w_frag<<<(tot + 255) / 256, 256, 0, stream>>>(Wl2, WPL, 64, tot);
    pack_w_frag<<<(tot + 255) / 256, 256, 0, stream>>>(Wr2, WPR, 64, tot);
  }
  node_gemm_dual<64><<<gemmBlocks, 128, 0, stream>>>(Y, WPL, WPR, XL, XR, N);
  fill_f32<<<(n4  + 255) / 256, 256, 0, stream>>>(M,  -INFINITY, n4);
  fill_f32<<<(n4  + 255) / 256, 256, 0, stream>>>(DEN, 0.0f, n4);
  fill_f32<<<(n64 + 255) / 256, 256, 0, stream>>>(ACC, 0.0f, n64);
  fill_f32<<<(64 * 64 + 64 + 255) / 256, 256, 0, stream>>>(GS, 0.0f, 64 * 64 + 64); // GS + GC contiguous
  edge_logits <<<edgeBlocks, 256, 0, stream>>>(src, dst, eattr, XL, XR, We2, at2, LG, M, E);
  edge_exp    <<<(eh + 255) / 256, 256, 0, stream>>>(dst, LG, M, DEN, eh);
  edge_scatter<<<edgeBlocks, 256, 0, stream>>>(src, dst, XL, LG, DEN, ACC, E);
  finalize_pool<<<(n64 + 255) / 256, 256, 0, stream>>>(ACC, b2, batch, GS, GC, n64);

  // ---------------- heads ----------------
  heads_kernel<<<1, 64, 0, stream>>>(GS, GC, Wg1, bg1, Wg2, bg2, Wp, bp,
                                     Wc1, bc1, Wc2, bc2, (float*)d_out);
}
